// GraphSAGEModel_6373731468068
// MI455X (gfx1250) — compile-verified, matrix operand order checked by reading, not run on hardware
//
#include <hip/hip_runtime.h>

#define NN 100000
#define NE 640000
#define IC 128
#define HC 256
#define OC 64

typedef __attribute__((ext_vector_type(16))) __bf16 v16bf;
typedef __attribute__((ext_vector_type(8)))  float  v8f;

__device__ __forceinline__ unsigned short f2bf(float f) {
    union { float f; unsigned int i; } c; c.f = f;
    unsigned int r = c.i + 0x7FFFu + ((c.i >> 16) & 1u);   // round-to-nearest-even
    return (unsigned short)(r >> 16);
}
__device__ __forceinline__ float bf2f(unsigned short u) {
    union { unsigned int i; float f; } c; c.i = ((unsigned int)u) << 16;
    return c.f;
}
__device__ __forceinline__ void atomAdd(float* p, float v) {
    unsafeAtomicAdd(p, v);   // native GLOBAL_ATOMIC_ADD_F32 on gfx1250
}

// ---------------- utility kernels ----------------
__global__ void k_zero_f32(float* __restrict__ p, int n) {
    for (int i = blockIdx.x * blockDim.x + threadIdx.x; i < n; i += gridDim.x * blockDim.x)
        p[i] = 0.0f;
}

__global__ void k_cvt_bf16(const float* __restrict__ a, unsigned short* __restrict__ b, int n) {
    for (int i = blockIdx.x * blockDim.x + threadIdx.x; i < n; i += gridDim.x * blockDim.x)
        b[i] = f2bf(a[i]);
}

__global__ void k_count(const int* __restrict__ dst, float* __restrict__ cnt, int e) {
    int t = blockIdx.x * blockDim.x + threadIdx.x;
    if (t < e) atomAdd(&cnt[dst[t]], 1.0f);
}

// edge-parallel scatter of f32 features, 4 channels/thread
__global__ void k_scatter_f32(const float* __restrict__ x, const int* __restrict__ src,
                              const int* __restrict__ dst, float* __restrict__ agg,
                              int e, int C) {
    int per = C >> 2;
    int t = blockIdx.x * blockDim.x + threadIdx.x;
    if (t >= e * per) return;
    int ed = t / per;
    int c  = (t - ed * per) * 4;
    int s = src[ed], d = dst[ed];
    const float4 v = *(const float4*)(x + (size_t)s * C + c);
    float* o = agg + (size_t)d * C + c;
    atomAdd(o + 0, v.x); atomAdd(o + 1, v.y);
    atomAdd(o + 2, v.z); atomAdd(o + 3, v.w);
}

// edge-parallel scatter of bf16 features, 8 channels/thread
__global__ void k_scatter_bf16(const unsigned short* __restrict__ xb, const int* __restrict__ src,
                               const int* __restrict__ dst, float* __restrict__ agg,
                               int e, int C) {
    int per = C >> 3;
    int t = blockIdx.x * blockDim.x + threadIdx.x;
    if (t >= e * per) return;
    int ed = t / per;
    int c  = (t - ed * per) * 8;
    int s = src[ed], d = dst[ed];
    const uint4 q = *(const uint4*)(xb + (size_t)s * C + c);
    float* o = agg + (size_t)d * C + c;
    unsigned int w[4] = { q.x, q.y, q.z, q.w };
#pragma unroll
    for (int i = 0; i < 4; i++) {
        atomAdd(o + 2 * i + 0, bf2f((unsigned short)(w[i] & 0xFFFFu)));
        atomAdd(o + 2 * i + 1, bf2f((unsigned short)(w[i] >> 16)));
    }
}

// mean = agg / max(cnt,1); emit bf16 (4 channels/thread)
__global__ void k_mean_cvt(const float* __restrict__ agg, const float* __restrict__ cnt,
                           unsigned short* __restrict__ mb, int nNodes, int C) {
    int per = C >> 2;
    int t = blockIdx.x * blockDim.x + threadIdx.x;
    if (t >= nNodes * per) return;
    int nd = t / per;
    float inv = 1.0f / fmaxf(cnt[nd], 1.0f);
    const float4 v = *(const float4*)(agg + (size_t)t * 4);
    unsigned int lo = (unsigned int)f2bf(v.x * inv) | ((unsigned int)f2bf(v.y * inv) << 16);
    unsigned int hi = (unsigned int)f2bf(v.z * inv) | ((unsigned int)f2bf(v.w * inv) << 16);
    *(uint2*)(mb + (size_t)t * 4) = make_uint2(lo, hi);
}

// ---------------- WMMA dual-GEMM: out = A1@B1^T + A2@B2^T + bias (opt. ReLU) ----------------
// ISA 16-bit A/B layout: lane L covers outer index (L&15); per K-step-of-32 the lane's
// 16 elements are K = half*8 + [0..7]  and  K = 16 + half*8 + [0..7]  (half = L>>4).
__device__ __forceinline__ v16bf load_op(const unsigned short* __restrict__ p, int kk, int half) {
    union { v16bf v; uint4 q[2]; } u;
    u.q[0] = *(const uint4*)(p + kk + half * 8);
    u.q[1] = *(const uint4*)(p + kk + 16 + half * 8);
    return u.v;
}

__global__ __launch_bounds__(256) void k_gemm(
    const unsigned short* __restrict__ A1, const unsigned short* __restrict__ A2,
    const unsigned short* __restrict__ B1, const unsigned short* __restrict__ B2,
    const float* __restrict__ bias, float* __restrict__ outf,
    unsigned short* __restrict__ outb, int M, int K, int outc, int relu) {
    const int lane = threadIdx.x;        // 0..31 (wave32)
    const int wave = threadIdx.y;        // 0..7
    const int gw = blockIdx.x * 8 + wave;
    const int ntiles = outc >> 4;
    const int mt = gw / ntiles;
    const int nt = gw - mt * ntiles;
    if (mt * 16 >= M) return;            // whole-wave uniform exit; EXEC all-ones for WMMA
    const int half = lane >> 4;
    const int idx  = lane & 15;
    const unsigned short* a1 = A1 + (size_t)(mt * 16 + idx) * K;
    const unsigned short* a2 = A2 + (size_t)(mt * 16 + idx) * K;
    const unsigned short* b1 = B1 + (size_t)(nt * 16 + idx) * K;  // weights [outc][K] row-major
    const unsigned short* b2 = B2 + (size_t)(nt * 16 + idx) * K;
    v8f acc = {};
    for (int kk = 0; kk < K; kk += 32) {
        v16bf av = load_op(a1, kk, half);
        v16bf bv = load_op(b1, kk, half);
        acc = __builtin_amdgcn_wmma_f32_16x16x32_bf16(false, av, false, bv,
                                                      (short)0, acc, false, false);
        av = load_op(a2, kk, half);
        bv = load_op(b2, kk, half);
        acc = __builtin_amdgcn_wmma_f32_16x16x32_bf16(false, av, false, bv,
                                                      (short)0, acc, false, false);
    }
    const float bb = bias[nt * 16 + idx];
#pragma unroll
    for (int r = 0; r < 8; r++) {        // row = r + half*8, col = idx (C/D VGPR layout)
        float v = acc[r] + bb;
        if (relu) v = fmaxf(v, 0.0f);
        size_t o = (size_t)(mt * 16 + half * 8 + r) * outc + nt * 16 + idx;
        if (outf) outf[o] = v;
        if (outb) outb[o] = f2bf(v);
    }
}

// ---------------- launch ----------------
extern "C" void kernel_launch(void* const* d_in, const int* in_sizes, int n_in,
                              void* d_out, int out_size, void* d_ws, size_t ws_size,
                              hipStream_t stream) {
    const float* x   = (const float*)d_in[0];
    const int*   ei  = (const int*)d_in[1];
    const float* W1l = (const float*)d_in[2];
    const float* b1  = (const float*)d_in[3];
    const float* W1r = (const float*)d_in[4];
    const float* W2l = (const float*)d_in[5];
    const float* b2  = (const float*)d_in[6];
    const float* W2r = (const float*)d_in[7];
    float* out = (float*)d_out;
    const int* src = ei;
    const int* dst = ei + NE;

    char* ws = (char*)d_ws;
    size_t off = 0;
    auto carve = [&](size_t bytes) { char* p = ws + off; off += (bytes + 255) & ~(size_t)255; return p; };
    float*          cnt  = (float*)         carve((size_t)NN * 4);
    float*          agg  = (float*)         carve((size_t)NN * HC * 4);
    unsigned short* xb   = (unsigned short*)carve((size_t)NN * IC * 2);
    unsigned short* hb   = (unsigned short*)carve((size_t)NN * HC * 2);
    unsigned short* mb   = (unsigned short*)carve((size_t)NN * HC * 2);
    unsigned short* w1lb = (unsigned short*)carve((size_t)HC * IC * 2);
    unsigned short* w1rb = (unsigned short*)carve((size_t)HC * IC * 2);
    unsigned short* w2lb = (unsigned short*)carve((size_t)OC * HC * 2);
    unsigned short* w2rb = (unsigned short*)carve((size_t)OC * HC * 2);

    const int B = 256;
    auto gsz = [](int n, int b) { return (n + b - 1) / b; };

    // bf16 stagings
    k_cvt_bf16<<<gsz(NN * IC, B), B, 0, stream>>>(x,   xb,   NN * IC);
    k_cvt_bf16<<<gsz(HC * IC, B), B, 0, stream>>>(W1l, w1lb, HC * IC);
    k_cvt_bf16<<<gsz(HC * IC, B), B, 0, stream>>>(W1r, w1rb, HC * IC);
    k_cvt_bf16<<<gsz(OC * HC, B), B, 0, stream>>>(W2l, w2lb, OC * HC);
    k_cvt_bf16<<<gsz(OC * HC, B), B, 0, stream>>>(W2r, w2rb, OC * HC);

    // degree counts (shared by both layers)
    k_zero_f32<<<2048, B, 0, stream>>>(cnt, NN);
    k_count<<<gsz(NE, B), B, 0, stream>>>(dst, cnt, NE);

    // ---- layer 1 ----
    k_zero_f32<<<4096, B, 0, stream>>>(agg, NN * IC);
    k_scatter_f32<<<gsz(NE * (IC / 4), B), B, 0, stream>>>(x, src, dst, agg, NE, IC);
    k_mean_cvt<<<gsz(NN * (IC / 4), B), B, 0, stream>>>(agg, cnt, mb, NN, IC);
    {   // 6250 m-tiles x 16 n-tiles = 100000 waves, 8 waves/block
        int waves = (NN / 16) * (HC / 16);
        k_gemm<<<dim3(gsz(waves, 8)), dim3(32, 8), 0, stream>>>(
            mb, xb, w1lb, w1rb, b1, nullptr, hb, NN, IC, HC, 1);
    }

    // ---- layer 2 ----
    k_zero_f32<<<8192, B, 0, stream>>>(agg, NN * HC);
    k_scatter_bf16<<<gsz(NE * (HC / 8), B), B, 0, stream>>>(hb, src, dst, agg, NE, HC);
    k_mean_cvt<<<gsz(NN * (HC / 4), B), B, 0, stream>>>(agg, cnt, mb, NN, HC);
    {   // 6250 x 4 = 25000 waves
        int waves = (NN / 16) * (OC / 16);
        k_gemm<<<dim3(gsz(waves, 8)), dim3(32, 8), 0, stream>>>(
            mb, hb, w2lb, w2rb, b2, out, nullptr, NN, HC, OC, 0);
    }
}
